// GraphProp_87101936763636
// MI455X (gfx1250) — matrix operand order repaired and checked by reading, not compiled, not run on hardware
//
#include <hip/hip_runtime.h>

// ---------------------------------------------------------------------------
// GGNN propagation on MI455X (gfx1250, wave32, WMMA, async LDS staging).
//
// Math restructure: msg GEMM over 1.6M edges is folded into node-level GEMMs:
//   a[n] = deg(n) * (hv[n] @ Wd.T + msg_b) + S[n] @ Ws.T,
//   S[n] = sum of hv[src] over incoming edges (raw-feature scatter).
// Edge phase: f32 atomics, ~1.6GB/round -> HBM-bound (~70us at 23.3TB/s).
// Node phase: bf16 WMMA 16x16x32 (f32 accum); weight B-fragments staged into
// LDS per block via global_load_async_to_lds_b128 (ASYNCcnt) so all 8 waves
// in a block share one L2 fetch and WMMAs feed from low-latency ds loads.
// ---------------------------------------------------------------------------

typedef __attribute__((ext_vector_type(16))) __bf16 v16bf;
typedef __attribute__((ext_vector_type(8)))  float  v8f;

#define NN 50000
#define EE 1600000
#define HH 128

static __device__ __forceinline__ float sigm_(float x) {
  return 1.0f / (1.0f + __expf(-x));
}
static __device__ __forceinline__ float tanh_(float x) {
  float e = __expf(2.0f * x);
  return 1.0f - 2.0f / (e + 1.0f);   // safe at +/-inf
}

static __device__ __forceinline__ v8f wmma_bf16(v16bf a, v16bf b, v8f c) {
  return __builtin_amdgcn_wmma_f32_16x16x32_bf16(
      false, a, false, b, (short)0, c, false, false);
}

// async copy 16B global -> LDS (gfx1250, tracked by ASYNCcnt)
static __device__ __forceinline__ void async_b128(unsigned lds_addr,
                                                  const void* gaddr) {
  asm volatile("global_load_async_to_lds_b128 %0, %1, off"
               :: "v"(lds_addr), "v"(gaddr) : "memory");
}
static __device__ __forceinline__ void wait_async0() {
  asm volatile("s_wait_asynccnt 0x0" ::: "memory");
}
static __device__ __forceinline__ unsigned lds_off(const void* p) {
  return (unsigned)(size_t)p;  // low 32 bits of LDS aperture addr = LDS offset
}

// ---- zero fill ------------------------------------------------------------
__global__ void zero_f32(float* __restrict__ p, long long n) {
  long long i = (long long)blockIdx.x * blockDim.x + threadIdx.x;
  if (i < n) p[i] = 0.0f;
}

// ---- repack weight W[Jmax, *] (used as B = W.T) into WMMA B-fragments -----
// B-frag (32Kx16N bf16): lane l holds n=l&15, khalf=l>>4; element i holds
// k_local = i + 16*khalf. Stored [frag][lane][16] contiguous (32B per lane).
__global__ void prep_wfrag(const float* __restrict__ W, __bf16* __restrict__ F,
                           int Jmax, int Kmax, int stride, int coloff) {
  int idx = blockIdx.x * blockDim.x + threadIdx.x;
  int total = Jmax * Kmax;
  if (idx >= total) return;
  int i    = idx & 15;
  int lane = (idx >> 4) & 31;
  int fid  = idx >> 9;
  int nK   = Kmax >> 5;
  int jt   = fid / nK;
  int kc   = fid - jt * nK;
  int j    = jt * 16 + (lane & 15);
  int k    = kc * 32 + i + ((lane >> 4) << 4);
  F[idx] = (__bf16)W[(size_t)j * stride + coloff + k];
}

// ---- repack activations X[nrows, Kmax] into WMMA A-fragments --------------
// A-frag (16Mx32K bf16): lane l holds m=l&15, khalf=l>>4; element i holds
// k_local = i + 16*khalf (contiguous, matches B-frag).
__global__ void prep_afrag(const float* __restrict__ X, __bf16* __restrict__ F,
                           int nrows, int Kmax) {
  int idx = blockIdx.x * blockDim.x + threadIdx.x;
  int total = nrows * Kmax;
  if (idx >= total) return;
  int i    = idx & 15;
  int lane = (idx >> 4) & 31;
  int fid  = idx >> 9;
  int nK   = Kmax >> 5;
  int mt   = fid / nK;
  int kc   = fid - mt * nK;
  int m    = mt * 16 + (lane & 15);
  int k    = kc * 32 + i + ((lane >> 4) << 4);
  F[idx] = (__bf16)X[(size_t)m * Kmax + k];
}

// ---- edge scatter: S[dst] += hv[src], deg[dst] += 1 -----------------------
__global__ void scatter_kernel(const float* __restrict__ hv,
                               const int* __restrict__ esrc,
                               const int* __restrict__ edst,
                               float* __restrict__ S, float* __restrict__ deg,
                               int nE) {
  int idx = blockIdx.x * blockDim.x + threadIdx.x;
  int e = idx >> 5;
  int l = idx & 31;
  if (e >= nE) return;
  int s = esrc[e];
  int d = edst[e];
  // prefetch a future edge's source row (global_prefetch_b8)
  int e2 = e + 128;
  if (e2 < nE) __builtin_prefetch(hv + (size_t)esrc[e2] * HH, 0, 1);
  const float* hs = hv + (size_t)s * HH;
  float* Sd = S + (size_t)d * HH;
#pragma unroll
  for (int i = 0; i < 4; ++i) {
    int k = l + 32 * i;
    atomicAdd(&Sd[k], hs[k]);
  }
  if (l == 0) atomicAdd(&deg[d], 1.0f);
}

// ---- GEMM1: a = deg * (Xh @ Wd.T + msg_b) + Xs @ Ws.T ---------------------
// one wave per 16-node tile; D layout: m = r + 8*(lane>>4), n = lane&15.
// Weight frags staged per block into LDS (async), shared by all 8 waves.
__global__ void __launch_bounds__(256) gemm1_kernel(
    const __bf16* __restrict__ Xh_f, const __bf16* __restrict__ Xs_f,
    const __bf16* __restrict__ Wd_f, const __bf16* __restrict__ Ws_f,
    const float* __restrict__ deg, const float* __restrict__ mb,
    float* __restrict__ a, int ntiles) {
  __shared__ __bf16 lw[8 * 512];  // frag 0..3: Wd kc, frag 4..7: Ws kc (8KB)
  int tid  = threadIdx.x;
  int lane = tid & 31;
  int wv   = (blockIdx.x * blockDim.x + tid) >> 5;
  bool active = wv < ntiles;
  int wave = active ? wv : (ntiles - 1);
  const __bf16* Ah = Xh_f + (size_t)wave * 4 * 512;
  const __bf16* As = Xs_f + (size_t)wave * 4 * 512;
  int lh = lane >> 4;
  float degv[8];
#pragma unroll
  for (int r = 0; r < 8; ++r) degv[r] = deg[wave * 16 + r + 8 * lh];
  for (int ct = 0; ct < 16; ++ct) {
    __syncthreads();  // previous iteration's LDS reads complete
#pragma unroll
    for (int c = tid; c < 512; c += 256) {  // 8 frags x 64 16B-chunks
      int f = c >> 6, o = c & 63;
      const __bf16* src =
          (f < 4) ? (Wd_f + ((size_t)(ct * 4 + f)) * 512)
                  : (Ws_f + ((size_t)(ct * 4 + (f - 4))) * 512);
      async_b128(lds_off(&lw[f * 512 + o * 8]), src + o * 8);
    }
    wait_async0();
    __syncthreads();
    v8f accD = {}, accS = {};
#pragma unroll
    for (int kc = 0; kc < 4; ++kc) {
      v16bf ah = *(const v16bf*)(Ah + kc * 512 + lane * 16);
      v16bf bd = *(const v16bf*)(&lw[kc * 512 + lane * 16]);
      accD = wmma_bf16(ah, bd, accD);
      v16bf as = *(const v16bf*)(As + kc * 512 + lane * 16);
      v16bf bs = *(const v16bf*)(&lw[(4 + kc) * 512 + lane * 16]);
      accS = wmma_bf16(as, bs, accS);
    }
    if (active) {
      int j = ct * 16 + (lane & 15);
      float b = mb[j];
#pragma unroll
      for (int r = 0; r < 8; ++r) {
        int row = wave * 16 + r + 8 * lh;
        a[(size_t)row * 256 + j] = degv[r] * (accD[r] + b) + accS[r];
      }
    }
  }
}

// ---- GEMM2 + GRU: gi = a@Wih.T+bih, gh = hv@Whh.T+bhh, gates, hv update --
// LDS frag map per ct: f in [0,24): Wih gate g=f>>3, kc=f&7;
//                      f in [24,36): Whh gate g=(f-24)>>2, kc=(f-24)&3.
__global__ void __launch_bounds__(256) gemm2_gru_kernel(
    const __bf16* __restrict__ a_f, const __bf16* __restrict__ Xh_f,
    const __bf16* __restrict__ Wih_f, const __bf16* __restrict__ Whh_f,
    const float* __restrict__ bih, const float* __restrict__ bhh,
    const float* __restrict__ hv_in, float* __restrict__ hv_out, int ntiles) {
  __shared__ __bf16 lw[36 * 512];  // 36KB staged weights per ct
  int tid  = threadIdx.x;
  int lane = tid & 31;
  int wv   = (blockIdx.x * blockDim.x + tid) >> 5;
  bool active = wv < ntiles;
  int wave = active ? wv : (ntiles - 1);
  const __bf16* Aa = a_f + (size_t)wave * 8 * 512;   // K = 256 -> 8 frags
  const __bf16* Ah = Xh_f + (size_t)wave * 4 * 512;  // K = 128 -> 4 frags
  int lh = lane >> 4;
  for (int ct = 0; ct < 8; ++ct) {  // 8 output col-tiles of H
    __syncthreads();
#pragma unroll
    for (int c = tid; c < 2304; c += 256) {  // 36 frags x 64 chunks, 9/thread
      int f = c >> 6, o = c & 63;
      const __bf16* src;
      if (f < 24) {
        int g = f >> 3, kc = f & 7;
        src = Wih_f + ((size_t)((g * 8 + ct) * 8 + kc)) * 512;
      } else {
        int f2 = f - 24, g = f2 >> 2, kc = f2 & 3;
        src = Whh_f + ((size_t)((g * 8 + ct) * 4 + kc)) * 512;
      }
      async_b128(lds_off(&lw[f * 512 + o * 8]), src + o * 8);
    }
    wait_async0();
    __syncthreads();
    v8f aR = {}, aZ = {}, aNi = {}, aNh = {};
#pragma unroll
    for (int kc = 0; kc < 8; ++kc) {
      v16bf av = *(const v16bf*)(Aa + kc * 512 + lane * 16);
      v16bf bR = *(const v16bf*)(&lw[(kc)      * 512 + lane * 16]);
      v16bf bZ = *(const v16bf*)(&lw[(8  + kc) * 512 + lane * 16]);
      v16bf bN = *(const v16bf*)(&lw[(16 + kc) * 512 + lane * 16]);
      aR  = wmma_bf16(av, bR, aR);
      aZ  = wmma_bf16(av, bZ, aZ);
      aNi = wmma_bf16(av, bN, aNi);
    }
#pragma unroll
    for (int kc = 0; kc < 4; ++kc) {
      v16bf hvv = *(const v16bf*)(Ah + kc * 512 + lane * 16);
      v16bf bR = *(const v16bf*)(&lw[(24 + kc)     * 512 + lane * 16]);
      v16bf bZ = *(const v16bf*)(&lw[(28 + kc)     * 512 + lane * 16]);
      v16bf bN = *(const v16bf*)(&lw[(32 + kc)     * 512 + lane * 16]);
      aR  = wmma_bf16(hvv, bR, aR);   // sigmoid(i_r + h_r): sum into same acc
      aZ  = wmma_bf16(hvv, bZ, aZ);
      aNh = wmma_bf16(hvv, bN, aNh);  // h_n kept separate (scaled by r)
    }
    if (active) {
      int j = ct * 16 + (lane & 15);
      float bR  = bih[j]       + bhh[j];
      float bZ  = bih[128 + j] + bhh[128 + j];
      float bNi = bih[256 + j];
      float bNh = bhh[256 + j];
#pragma unroll
      for (int r = 0; r < 8; ++r) {
        int row = wave * 16 + r + 8 * lh;
        float rg = sigm_(aR[r] + bR);
        float zg = sigm_(aZ[r] + bZ);
        float ng = tanh_(aNi[r] + bNi + rg * (aNh[r] + bNh));
        float h  = hv_in[(size_t)row * HH + j];
        hv_out[(size_t)row * HH + j] = (1.0f - zg) * ng + zg * h;
      }
    }
  }
}

// ---------------------------------------------------------------------------
extern "C" void kernel_launch(void* const* d_in, const int* in_sizes, int n_in,
                              void* d_out, int out_size, void* d_ws, size_t ws_size,
                              hipStream_t stream) {
  const float* hv0  = (const float*)d_in[0];
  const int*   esrc = (const int*)d_in[1];
  const int*   edst = (const int*)d_in[2];
  const float* msgW = (const float*)d_in[3];   // [2,256,256]
  const float* msgb = (const float*)d_in[4];   // [2,256]
  const float* Wih  = (const float*)d_in[5];   // [2,384,256]
  const float* Whh  = (const float*)d_in[6];   // [2,384,128]
  const float* bih  = (const float*)d_in[7];   // [2,384]
  const float* bhh  = (const float*)d_in[8];   // [2,384]
  float* out = (float*)d_out;

  // workspace carve-out (256B aligned)
  char* p = (char*)d_ws;
  auto carve = [&](size_t bytes) -> void* {
    void* r = (void*)p;
    p += (bytes + 255) & ~(size_t)255;
    return r;
  };
  float* S      = (float*)carve((size_t)NN * HH * 4 + (size_t)NN * 4);
  float* deg    = S + (size_t)NN * HH;                  // contiguous after S
  float* hv_tmp = (float*)carve((size_t)NN * HH * 4);
  float* a      = (float*)carve((size_t)NN * 256 * 4);
  __bf16* Xh_f  = (__bf16*)carve((size_t)NN * HH * 2);
  __bf16* Xs_f  = (__bf16*)carve((size_t)NN * HH * 2);
  __bf16* a_f   = (__bf16*)carve((size_t)NN * 256 * 2);
  __bf16* Wd_f  = (__bf16*)carve((size_t)256 * 128 * 2);
  __bf16* Ws_f  = (__bf16*)carve((size_t)256 * 128 * 2);
  __bf16* Wih_f = (__bf16*)carve((size_t)384 * 256 * 2);
  __bf16* Whh_f = (__bf16*)carve((size_t)384 * 128 * 2);

  const int ntiles = NN / 16;  // 3125
  auto cdiv = [](long long x, long long y) { return (int)((x + y - 1) / y); };
  dim3 blk(256);

  for (int t = 0; t < 2; ++t) {
    const float* hv_in  = (t == 0) ? hv0 : hv_tmp;
    float*       hv_out = (t == 0) ? hv_tmp : out;
    const float* mW    = msgW + (size_t)t * 256 * 256;
    const float* mb    = msgb + (size_t)t * 256;
    const float* Wih_t = Wih  + (size_t)t * 384 * 256;
    const float* Whh_t = Whh  + (size_t)t * 384 * 128;
    const float* bih_t = bih  + (size_t)t * 384;
    const float* bhh_t = bhh  + (size_t)t * 384;

    long long zn = (long long)NN * HH + NN;  // S + deg
    zero_f32<<<cdiv(zn, 256), blk, 0, stream>>>(S, zn);

    prep_wfrag<<<cdiv(256 * 128, 256), blk, 0, stream>>>(mW,    Wd_f,  256, 128, 256, 0);
    prep_wfrag<<<cdiv(256 * 128, 256), blk, 0, stream>>>(mW,    Ws_f,  256, 128, 256, 128);
    prep_wfrag<<<cdiv(384 * 256, 256), blk, 0, stream>>>(Wih_t, Wih_f, 384, 256, 256, 0);
    prep_wfrag<<<cdiv(384 * 128, 256), blk, 0, stream>>>(Whh_t, Whh_f, 384, 128, 128, 0);

    prep_afrag<<<cdiv((long long)NN * HH, 256), blk, 0, stream>>>(hv_in, Xh_f, NN, HH);

    scatter_kernel<<<cdiv((long long)EE * 32, 256), blk, 0, stream>>>(
        hv_in, esrc, edst, S, deg, EE);

    prep_afrag<<<cdiv((long long)NN * HH, 256), blk, 0, stream>>>(S, Xs_f, NN, HH);

    gemm1_kernel<<<cdiv((long long)ntiles * 32, 256), blk, 0, stream>>>(
        Xh_f, Xs_f, Wd_f, Ws_f, deg, mb, a, ntiles);

    prep_afrag<<<cdiv((long long)NN * 256, 256), blk, 0, stream>>>(a, a_f, NN, 256);

    gemm2_gru_kernel<<<cdiv((long long)ntiles * 32, 256), blk, 0, stream>>>(
        a_f, Xh_f, Wih_f, Whh_f, bih_t, bhh_t, hv_in, hv_out, ntiles);
  }
}
